// ImprovedPointTransformer_61478161875084
// MI455X (gfx1250) — matrix operand order repaired and checked by reading, not compiled, not run on hardware
//
#include <hip/hip_runtime.h>
#include <hip/hip_bf16.h>
#include <math.h>

// =====================================================================
// ImprovedPointTransformer forward for MI455X (gfx1250, wave32, WMMA).
//
// GEMMs: v_wmma_f32_16x16x32_f16, one wave per 16x64 output strip
// (4 WMMAs per K-step sharing one A fragment). Weights pre-packed to
// f16 [Npad64][Kpad32] (transposed, zero padded) => B fragments are
// guard-free global_load_b128 pairs. A tiles staged to LDS coalesced;
// edge tiles use branchless clamped loads (no EXEC-mask serialization).
//
// Param order assumption: jax tree_flatten (sorted dict keys) =>
//   d1,d2,d3,mbn1,mbn2,mlp1,mlp2,mlp3,t1..t7,u1,u2,u3, then x.
//   lin_p leaves: b,w.  bn_p leaves: b,g.
//   ptl_p leaves: bn1,bn2,delta1,delta2,fc1,fc2,gamma1,gamma2,wk,wq,wv.
// =====================================================================

typedef __attribute__((ext_vector_type(16))) _Float16 v16h;
typedef __attribute__((ext_vector_type(8)))  _Float16 v8h;
typedef __attribute__((ext_vector_type(8)))  float    v8f;

#define KNN   16
#define BATCH 4
#define NT    4        // N-tiles per wave (16x64 strip)

static inline int ceilDiv(int a, int b) { return (a + b - 1) / b; }

// ---------------------------------------------------------------------
// Weight pack: W[K,N] f32 -> Wt[Npad][Kpad] f16 (transposed, padded).
// ---------------------------------------------------------------------
__global__ void pack_w_kernel(const float* __restrict__ W, _Float16* __restrict__ Wt,
                              int Kd, int N, int Kpad, int Npad)
{
    int i = blockIdx.x * blockDim.x + threadIdx.x;
    if (i >= Npad * Kpad) return;
    int k = i % Kpad;
    int n = i / Kpad;
    Wt[i] = (k < Kd && n < N) ? (_Float16)W[(size_t)k * N + n] : (_Float16)0.f;
}

// ---------------------------------------------------------------------
// GEMM: C[M,N] = act(A[M,K] @ W[K,N] + bias).
// One wave computes a 16x64 strip: 4 accumulators share one A fragment.
// ---------------------------------------------------------------------
__global__ void gemm_wmma_kernel(const float* __restrict__ A,
                                 const _Float16* __restrict__ Wt,  // [Npad][Kpad]
                                 const float* __restrict__ bias,
                                 float* __restrict__ C,
                                 int M, int Kd, int N, int Kpad, int relu)
{
    __shared__ _Float16 As[16][32];            // [m][k], rows 64B apart
    const int lane = threadIdx.x;              // 0..31 (full wave)
    const int m0 = blockIdx.x * 16;
    const int n0 = blockIdx.y * (16 * NT);
    const int mr = lane & 15;                  // A row / B column for this lane
    const int kg = lane >> 4;                  // lane-group K offset select

    // B column base pointers; Wt rows padded to Npad (mult of 64): always safe.
    const _Float16* wr[NT];
    #pragma unroll
    for (int j = 0; j < NT; ++j)
        wr[j] = Wt + (size_t)(n0 + j * 16 + mr) * Kpad;

    const bool fullM = (m0 + 16 <= M);
    const int kt = (Kd + 31) / 32;

    v8f acc[NT];
    #pragma unroll
    for (int j = 0; j < NT; ++j) acc[j] = (v8f){};

    for (int t = 0; t < kt; ++t) {
        const int k0 = t * 32;
        // ---- stage A tile (16x32) in LDS, coalesced: row uniform, lane = K.
        if (fullM && (k0 + 32 <= Kd)) {
            #pragma unroll
            for (int i = 0; i < 16; ++i)
                As[i][lane] = (_Float16)A[(size_t)(m0 + i) * Kd + k0 + lane];
        } else {
            // Branchless edge path: clamp address (always valid), mask value.
            #pragma unroll
            for (int i = 0; i < 16; ++i) {
                int gm = m0 + i, gk = k0 + lane;
                int cm = gm < M ? gm : (M - 1);
                int ck = gk < Kd ? gk : (Kd - 1);
                float v = A[(size_t)cm * Kd + ck];
                As[i][lane] = (gm < M && gk < Kd) ? (_Float16)v : (_Float16)0.f;
            }
        }
        // Prefetch next A K-tile (global_prefetch_b8, speculative-safe).
        if (t + 1 < kt && fullM && (k0 + 32 + lane) < Kd)
            __builtin_prefetch(&A[(size_t)(m0 + mr) * Kd + k0 + 32], 0, 1);
        __syncthreads();

        // ---- fragments (contiguous K order, same for A and B) ----
        // A lane fragment = 16 contiguous halves: K = kg*16..kg*16+15
        v8h alo = *(const v8h*)&As[mr][kg * 16];
        v8h ahi = *(const v8h*)&As[mr][kg * 16 + 8];
        v16h a = __builtin_shufflevector(alo, ahi,
                    0, 1, 2, 3, 4, 5, 6, 7, 8, 9, 10, 11, 12, 13, 14, 15);
        // 4 B fragments (16 contiguous halves each) -> 4 WMMAs on one A frag.
        #pragma unroll
        for (int j = 0; j < NT; ++j) {
            const v8h* bp = (const v8h*)(wr[j] + k0 + kg * 16);
            v8h b0 = bp[0], b1 = bp[1];
            v16h b = __builtin_shufflevector(b0, b1,
                        0, 1, 2, 3, 4, 5, 6, 7, 8, 9, 10, 11, 12, 13, 14, 15);
            acc[j] = __builtin_amdgcn_wmma_f32_16x16x32_f16(
                        false, a, false, b, (short)0, acc[j], false, false);
        }
        __syncthreads();
    }

    // C/D layout: col = lane&15; row = r + 8*(lane>=16).
    const int hi = lane >> 4;
    #pragma unroll
    for (int j = 0; j < NT; ++j) {
        int gn = n0 + j * 16 + mr;
        float bv = (bias != nullptr && gn < N) ? bias[gn] : 0.f;
        #pragma unroll
        for (int r = 0; r < 8; ++r) {
            int gm = m0 + r + 8 * hi;
            if (gm < M && gn < N) {
                float v = acc[j][r] + bv;
                if (relu) v = fmaxf(v, 0.f);
                C[(size_t)gm * N + gn] = v;
            }
        }
    }
}

// ---------------------------------------------------------------------
// kNN: per-query register insertion sort (smallest KSEL squared dists).
// ---------------------------------------------------------------------
template <int KSEL>
__global__ void knn_kernel(const float* __restrict__ Q, const float* __restrict__ P,
                           int Nq, int Np, int* __restrict__ oIdx,
                           float* __restrict__ oDist)
{
    int n = blockIdx.x * blockDim.x + threadIdx.x;
    int b = blockIdx.y;
    if (n >= Nq) return;
    const float* q = Q + ((size_t)b * Nq + n) * 3;
    const float qx = q[0], qy = q[1], qz = q[2];
    const float* pb = P + (size_t)b * Np * 3;

    float bd[KSEL];
    int   bi[KSEL];
    #pragma unroll
    for (int t = 0; t < KSEL; ++t) { bd[t] = 3.0e38f; bi[t] = 0; }

    for (int j = 0; j < Np; ++j) {
        float dx = qx - pb[j * 3 + 0];
        float dy = qy - pb[j * 3 + 1];
        float dz = qz - pb[j * 3 + 2];
        float d = dx * dx + dy * dy + dz * dz;
        if (d < bd[KSEL - 1]) {
            bd[KSEL - 1] = d; bi[KSEL - 1] = j;
            #pragma unroll
            for (int t = KSEL - 1; t > 0; --t) {
                if (bd[t] < bd[t - 1]) {
                    float td = bd[t]; bd[t] = bd[t - 1]; bd[t - 1] = td;
                    int   ti = bi[t]; bi[t] = bi[t - 1]; bi[t - 1] = ti;
                }
            }
        }
    }
    size_t base = ((size_t)b * Nq + n) * KSEL;
    #pragma unroll
    for (int t = 0; t < KSEL; ++t) {
        oIdx[base + t] = bi[t];
        if (oDist) oDist[base + t] = bd[t];
    }
}

// ---------------------------------------------------------------------
// BatchNorm: population stats per channel + fused apply.
// ---------------------------------------------------------------------
__global__ void bn_stats_kernel(const float* __restrict__ X, int M, int C,
                                float* __restrict__ mean, float* __restrict__ var)
{
    __shared__ float ss[256];
    __shared__ float sq[256];
    const int c = blockIdx.x;
    const int tid = threadIdx.x;
    float s = 0.f, q = 0.f;
    for (int i = tid; i < M; i += blockDim.x) {
        float v = X[(size_t)i * C + c];
        s += v; q += v * v;
    }
    ss[tid] = s; sq[tid] = q;
    __syncthreads();
    for (int st = 128; st > 0; st >>= 1) {
        if (tid < st) { ss[tid] += ss[tid + st]; sq[tid] += sq[tid + st]; }
        __syncthreads();
    }
    if (tid == 0) {
        float m = ss[0] / (float)M;
        mean[c] = m;
        var[c]  = sq[0] / (float)M - m * m;
    }
}

__global__ void bn_apply_kernel(const float* __restrict__ X,
                                const float* __restrict__ mean,
                                const float* __restrict__ var,
                                const float* __restrict__ g,
                                const float* __restrict__ bb,
                                const float* __restrict__ resid,
                                float* __restrict__ Y,
                                int total, int C, int relu)
{
    int i = blockIdx.x * blockDim.x + threadIdx.x;
    if (i >= total) return;
    int c = i % C;
    float y = (X[i] - mean[c]) * rsqrtf(var[c] + 1e-5f) * g[c] + bb[c];
    if (resid) y += resid[i];
    if (relu) y = fmaxf(y, 0.f);
    Y[i] = y;
}

// ---------------------------------------------------------------------
// Elementwise / gather / attention kernels
// ---------------------------------------------------------------------
__global__ void gather_kernel(const float* __restrict__ X, const int* __restrict__ idx,
                              float* __restrict__ Y, int total,
                              int Np, int Nq, int C, int Kk)
{
    int i = blockIdx.x * blockDim.x + threadIdx.x;
    if (i >= total) return;
    int c = i % C;
    int r = i / C;                       // b*Nq*Kk + n*Kk + k
    int b = r / (Nq * Kk);
    Y[i] = X[((size_t)b * Np + idx[r]) * C + c];
}

__global__ void relpos_kernel(const float* __restrict__ xyz, const int* __restrict__ idx,
                              float* __restrict__ rel, int N, int total)
{
    int t = blockIdx.x * blockDim.x + threadIdx.x;
    if (t >= total) return;                     // total = B*N*KNN
    int n = (t / KNN) % N;
    int b = t / (KNN * N);
    int j = idx[t];
    const float* a = xyz + ((size_t)b * N + n) * 3;
    const float* p = xyz + ((size_t)b * N + j) * 3;
    rel[(size_t)t * 3 + 0] = a[0] - p[0];
    rel[(size_t)t * 3 + 1] = a[1] - p[1];
    rel[(size_t)t * 3 + 2] = a[2] - p[2];
}

// In-place: T[b,n,k,c] = Q[b,n,c] - T[b,n,k,c] + P[b,n,k,c]
__global__ void qkp_kernel(const float* __restrict__ Q, float* __restrict__ T,
                           const float* __restrict__ P, int total, int C)
{
    int i = blockIdx.x * blockDim.x + threadIdx.x;
    if (i >= total) return;
    int c = i % C;
    int bn = i / (C * KNN);
    T[i] = Q[(size_t)bn * C + c] - T[i] + P[i];
}

// Per (b,n,c): softmax over K of logits/sqrt(C); agg = sum p*(V+P).
__global__ void attn_kernel(const float* __restrict__ L, const float* __restrict__ V,
                            const float* __restrict__ P, float* __restrict__ O,
                            int total, int C)
{
    int i = blockIdx.x * blockDim.x + threadIdx.x;
    if (i >= total) return;
    int c = i % C;
    int bn = i / C;
    size_t base = (size_t)bn * KNN * C + c;
    float sc = rsqrtf((float)C);
    float mx = -3.0e38f;
    #pragma unroll
    for (int k = 0; k < KNN; ++k)
        mx = fmaxf(mx, L[base + (size_t)k * C] * sc);
    float s = 0.f, acc = 0.f;
    #pragma unroll
    for (int k = 0; k < KNN; ++k) {
        size_t o = base + (size_t)k * C;
        float e = __expf(L[o] * sc - mx);
        s += e;
        acc += e * (V[o] + P[o]);
    }
    O[i] = acc / s;
}

__global__ void maxk_kernel(const float* __restrict__ X, float* __restrict__ Y,
                            int total, int C)
{
    int i = blockIdx.x * blockDim.x + threadIdx.x;
    if (i >= total) return;
    int c = i % C;
    int bn = i / C;
    size_t base = (size_t)bn * KNN * C + c;
    float m = -3.0e38f;
    #pragma unroll
    for (int k = 0; k < KNN; ++k)
        m = fmaxf(m, X[base + (size_t)k * C]);
    Y[i] = m;
}

__global__ void halve_xyz_kernel(const float* __restrict__ xyz, float* __restrict__ o, int N)
{
    int i = blockIdx.x * blockDim.x + threadIdx.x;
    int Nh = N / 2;
    if (i >= BATCH * Nh * 3) return;
    int d = i % 3;
    int n = (i / 3) % Nh;
    int b = i / (3 * Nh);
    o[i] = xyz[((size_t)b * N + n) * 3 + d];
}

// 3-NN inverse-distance interpolation: w = 1/(d+1e-8), normalized.
__global__ void interp3_kernel(const float* __restrict__ X2, const int* __restrict__ idx,
                               const float* __restrict__ dist, float* __restrict__ Y,
                               int total, int C, int Nq, int Np)
{
    int i = blockIdx.x * blockDim.x + threadIdx.x;
    if (i >= total) return;
    int c = i % C;
    int r = i / C;                  // b*Nq + n
    int b = r / Nq;
    const int*   id = idx  + (size_t)r * 3;
    const float* dd = dist + (size_t)r * 3;
    float w0 = 1.f / (dd[0] + 1e-8f);
    float w1 = 1.f / (dd[1] + 1e-8f);
    float w2 = 1.f / (dd[2] + 1e-8f);
    float ws = w0 + w1 + w2;
    const float* xb = X2 + (size_t)b * Np * C;
    Y[i] = (w0 * xb[(size_t)id[0] * C + c] +
            w1 * xb[(size_t)id[1] * C + c] +
            w2 * xb[(size_t)id[2] * C + c]) / ws;
}

// g[b,:] = concat(max_n f4, max_n f3, max_n f2, max_n u1)  -> (4,960)
__global__ void concat_globalmax_kernel(const float* __restrict__ f4,
                                        const float* __restrict__ f3,
                                        const float* __restrict__ f2,
                                        const float* __restrict__ u1,
                                        float* __restrict__ G)
{
    int i = blockIdx.x * blockDim.x + threadIdx.x;
    if (i >= BATCH * 960) return;
    int c = i % 960;
    int b = i / 960;
    const float* src; int C, N, cl;
    if (c < 512)      { src = f4; C = 512; N = 512;  cl = c; }
    else if (c < 768) { src = f3; C = 256; N = 1024; cl = c - 512; }
    else if (c < 896) { src = f2; C = 128; N = 2048; cl = c - 768; }
    else              { src = u1; C = 64;  N = 4096; cl = c - 896; }
    const float* p = src + (size_t)b * N * C + cl;
    float m = -3.0e38f;
    for (int n = 0; n < N; ++n) m = fmaxf(m, p[(size_t)n * C]);
    G[i] = m;
}

// =====================================================================
// Host-side orchestration
// =====================================================================
struct LinP { const float* w; const float* b; };
struct BnP  { const float* g; const float* b; };
struct TdP  { BnP bn1; LinP fc1; };
struct PtlPar {
    BnP bn1, bn2;
    LinP delta1, delta2, fc1, fc2, gamma1, gamma2;
    LinP wk, wq, wv;
};

struct Cur {
    void* const* in; int i;
    const float* n() { return (const float*)in[i++]; }
};
static BnP rdBn(Cur& c)  { BnP p; p.b = c.n(); p.g = c.n(); return p; }  // sorted: b,g
static LinP rdLin(Cur& c, bool bias) {
    LinP p; p.b = bias ? c.n() : nullptr; p.w = c.n(); return p;          // sorted: b,w
}
static TdP rdTd(Cur& c)  { TdP t; t.bn1 = rdBn(c); t.fc1 = rdLin(c, true); return t; }
static PtlPar rdPtl(Cur& c) {
    PtlPar p;
    p.bn1 = rdBn(c); p.bn2 = rdBn(c);
    p.delta1 = rdLin(c, true); p.delta2 = rdLin(c, true);
    p.fc1 = rdLin(c, true);    p.fc2 = rdLin(c, true);
    p.gamma1 = rdLin(c, true); p.gamma2 = rdLin(c, true);
    p.wk = rdLin(c, false); p.wq = rdLin(c, false); p.wv = rdLin(c, false);
    return p;
}

struct Ws {
    float *xyz2, *xyz3, *xyz4;
    float *f1, *f2, *f3, *f4, *u3f, *u2f, *u1f, *featbuf;
    float *xbuf, *qbuf, *tmp1, *tmp2, *aggbuf, *relbuf;
    float *bufA, *bufB, *bufC, *bufD;
    float *meanb, *varb, *gbuf, *h1, *h2, *dist3;
    _Float16 *wt;
    int *knn, *knn3;
};

static void gemm(hipStream_t s, const float* A, const LinP& L, float* C,
                 int M, int K, int N, int relu, Ws& w)
{
    const int Kpad = ceilDiv(K, 32) * 32;
    const int Npad = ceilDiv(N, 16 * NT) * (16 * NT);   // pad to strip width
    pack_w_kernel<<<ceilDiv(Npad * Kpad, 256), 256, 0, s>>>(L.w, w.wt, K, N, Kpad, Npad);
    dim3 g((unsigned)ceilDiv(M, 16), (unsigned)(Npad / (16 * NT)));
    gemm_wmma_kernel<<<g, 32, 0, s>>>(A, w.wt, L.b, C, M, K, N, Kpad, relu);
}

static void bnrun(hipStream_t s, const float* X, const BnP& B, float* Y,
                  int M, int C, int relu, const float* resid, Ws& w)
{
    bn_stats_kernel<<<C, 256, 0, s>>>(X, M, C, w.meanb, w.varb);
    int tot = M * C;
    bn_apply_kernel<<<ceilDiv(tot, 256), 256, 0, s>>>(
        X, w.meanb, w.varb, B.g, B.b, resid, Y, tot, C, relu);
}

static void run_ptl(hipStream_t s, const float* xyz, const float* feats,
                    int dp, int dm, int N, const PtlPar& P, float* out, Ws& w)
{
    const int M1 = BATCH * N;
    const int MK = M1 * KNN;
    // x = relu(bn1(fc1(feats)))
    gemm(s, feats, P.fc1, w.tmp1, M1, dp, dm, 0, w);
    bnrun(s, w.tmp1, P.bn1, w.xbuf, M1, dm, 1, nullptr, w);
    // kNN + relative positions
    knn_kernel<16><<<dim3(ceilDiv(N, 128), BATCH), 128, 0, s>>>(
        xyz, xyz, N, N, w.knn, nullptr);
    relpos_kernel<<<ceilDiv(MK, 256), 256, 0, s>>>(xyz, w.knn, w.relbuf, N, MK);
    // pos = delta2(relu(delta1(rel)))
    gemm(s, w.relbuf, P.delta1, w.bufA, MK, 3, dm, 1, w);
    gemm(s, w.bufA,   P.delta2, w.bufB, MK, dm, dm, 0, w);   // pos -> bufB
    // q, k(gathered), v(gathered)
    gemm(s, w.xbuf, P.wq, w.qbuf, M1, dm, dm, 0, w);
    gemm(s, w.xbuf, P.wk, w.tmp1, M1, dm, dm, 0, w);
    gather_kernel<<<ceilDiv(MK * dm, 256), 256, 0, s>>>(
        w.tmp1, w.knn, w.bufA, MK * dm, N, N, dm, KNN);      // k -> bufA
    gemm(s, w.xbuf, P.wv, w.tmp1, M1, dm, dm, 0, w);
    gather_kernel<<<ceilDiv(MK * dm, 256), 256, 0, s>>>(
        w.tmp1, w.knn, w.bufC, MK * dm, N, N, dm, KNN);      // v -> bufC
    // t = q - k + pos  (in place on bufA)
    qkp_kernel<<<ceilDiv(MK * dm, 256), 256, 0, s>>>(
        w.qbuf, w.bufA, w.bufB, MK * dm, dm);
    // a = gamma2(relu(gamma1(t)))
    gemm(s, w.bufA, P.gamma1, w.bufD, MK, dm, dm, 1, w);
    gemm(s, w.bufD, P.gamma2, w.bufA, MK, dm, dm, 0, w);     // logits -> bufA
    // softmax over K + aggregate with (v + pos)
    attn_kernel<<<ceilDiv(M1 * dm, 256), 256, 0, s>>>(
        w.bufA, w.bufC, w.bufB, w.aggbuf, M1 * dm, dm);
    // out = relu(bn2(fc2(agg)) + x)
    gemm(s, w.aggbuf, P.fc2, w.tmp1, M1, dm, dm, 0, w);
    bnrun(s, w.tmp1, P.bn2, out, M1, dm, 1, w.xbuf, w);
}

static void run_down(hipStream_t s, const float* x, const float* xyz,
                     int N, int Ci, int Co, const TdP& T,
                     float* outFeat, float* outXyz, Ws& w)
{
    const int Nh = N / 2;
    halve_xyz_kernel<<<ceilDiv(BATCH * Nh * 3, 256), 256, 0, s>>>(xyz, outXyz, N);
    knn_kernel<16><<<dim3(ceilDiv(Nh, 128), BATCH), 128, 0, s>>>(
        outXyz, xyz, Nh, N, w.knn, nullptr);
    const int MK = BATCH * Nh * KNN;
    gather_kernel<<<ceilDiv(MK * Ci, 256), 256, 0, s>>>(
        x, w.knn, w.bufA, MK * Ci, N, Nh, Ci, KNN);
    gemm(s, w.bufA, T.fc1, w.bufB, MK, Ci, Co, 0, w);
    bnrun(s, w.bufB, T.bn1, w.bufB, MK, Co, 1, nullptr, w);
    maxk_kernel<<<ceilDiv(BATCH * Nh * Co, 256), 256, 0, s>>>(
        w.bufB, outFeat, BATCH * Nh * Co, Co);
}

static void run_up(hipStream_t s, const float* x2, const float* xyz1, const float* xyz2,
                   int N1, int N2, int Ci, int Co, const TdP& U, float* outFeat, Ws& w)
{
    knn_kernel<3><<<dim3(ceilDiv(N1, 128), BATCH), 128, 0, s>>>(
        xyz1, xyz2, N1, N2, w.knn3, w.dist3);
    interp3_kernel<<<ceilDiv(BATCH * N1 * Ci, 256), 256, 0, s>>>(
        x2, w.knn3, w.dist3, w.tmp2, BATCH * N1 * Ci, Ci, N1, N2);
    gemm(s, w.tmp2, U.fc1, w.tmp1, BATCH * N1, Ci, Co, 0, w);
    bnrun(s, w.tmp1, U.bn1, outFeat, BATCH * N1, Co, 1, nullptr, w);
}

extern "C" void kernel_launch(void* const* d_in, const int* in_sizes, int n_in,
                              void* d_out, int out_size, void* d_ws, size_t ws_size,
                              hipStream_t stream)
{
    (void)in_sizes; (void)n_in; (void)out_size;

    // ---- parameters (sorted-key pytree order) ----
    Cur c{d_in, 0};
    TdP d1 = rdTd(c), d2 = rdTd(c), d3 = rdTd(c);
    BnP mbn1 = rdBn(c), mbn2 = rdBn(c);
    LinP mlp1 = rdLin(c, true), mlp2 = rdLin(c, true), mlp3 = rdLin(c, true);
    PtlPar t1 = rdPtl(c), t2 = rdPtl(c), t3 = rdPtl(c), t4 = rdPtl(c),
           t5 = rdPtl(c), t6 = rdPtl(c), t7 = rdPtl(c);
    TdP u1 = rdTd(c), u2 = rdTd(c), u3 = rdTd(c);
    const float* x = c.n();               // (4,4096,3), also xyz level 1

    // ---- workspace bump allocator ----
    char* base = (char*)d_ws;
    size_t off = 0;
    auto allocf = [&](size_t n) -> float* {
        float* p = (float*)(base + off);
        off += ((n * sizeof(float) + 255) / 256) * 256;
        return p;
    };
    auto alloci = [&](size_t n) -> int* {
        int* p = (int*)(base + off);
        off += ((n * sizeof(int) + 255) / 256) * 256;
        return p;
    };

    Ws w;
    const size_t L = 1048576;             // B*N*dm is 1M floats at every level
    w.xyz2 = allocf(BATCH * 2048 * 3);
    w.xyz3 = allocf(BATCH * 1024 * 3);
    w.xyz4 = allocf(BATCH * 512 * 3);
    w.f1 = allocf(L); w.f2 = allocf(L); w.f3 = allocf(L); w.f4 = allocf(L);
    w.u3f = allocf(L); w.u2f = allocf(L); w.u1f = allocf(L);
    w.featbuf = allocf(L);
    w.xbuf = allocf(L); w.qbuf = allocf(L); w.tmp1 = allocf(L);
    w.tmp2 = allocf(2 * L);               // interp output up to (4,1024,512)
    w.aggbuf = allocf(L);
    w.relbuf = allocf((size_t)BATCH * 4096 * KNN * 3);
    w.bufA = allocf(16 * L); w.bufB = allocf(16 * L);
    w.bufC = allocf(16 * L); w.bufD = allocf(16 * L);
    w.meanb = allocf(512); w.varb = allocf(512);
    w.gbuf = allocf(BATCH * 960);
    w.h1 = allocf(BATCH * 512); w.h2 = allocf(BATCH * 256);
    w.dist3 = allocf((size_t)BATCH * 4096 * 3);
    w.wt = (_Float16*)allocf(524288);     // packed weights: up to 1M halves
    w.knn  = alloci((size_t)BATCH * 4096 * KNN);
    w.knn3 = alloci((size_t)BATCH * 4096 * 3);
    if (off > ws_size) return;            // scratch too small: bail deterministically

    // ---- encoder ----
    run_ptl(stream, x, x, 3, 64, 4096, t1, w.f1, w);
    run_down(stream, w.f1, x, 4096, 64, 128, d1, w.featbuf, w.xyz2, w);
    run_ptl(stream, w.xyz2, w.featbuf, 128, 128, 2048, t2, w.f2, w);
    run_down(stream, w.f2, w.xyz2, 2048, 128, 256, d2, w.featbuf, w.xyz3, w);
    run_ptl(stream, w.xyz3, w.featbuf, 256, 256, 1024, t3, w.f3, w);
    run_down(stream, w.f3, w.xyz3, 1024, 256, 512, d3, w.featbuf, w.xyz4, w);
    run_ptl(stream, w.xyz4, w.featbuf, 512, 512, 512, t4, w.f4, w);

    // ---- decoder ----
    run_up(stream, w.f4, w.xyz3, w.xyz4, 1024, 512, 512, 256, u1, w.featbuf, w);
    run_ptl(stream, w.xyz3, w.featbuf, 256, 256, 1024, t5, w.u3f, w);
    run_up(stream, w.u3f, w.xyz2, w.xyz3, 2048, 1024, 256, 128, u2, w.featbuf, w);
    run_ptl(stream, w.xyz2, w.featbuf, 128, 128, 2048, t6, w.u2f, w);
    run_up(stream, w.u2f, x, w.xyz2, 4096, 2048, 128, 64, u3, w.featbuf, w);
    run_ptl(stream, x, w.featbuf, 64, 64, 4096, t7, w.u1f, w);

    // ---- global pooling + MLP head ----
    concat_globalmax_kernel<<<ceilDiv(BATCH * 960, 256), 256, 0, stream>>>(
        w.f4, w.f3, w.f2, w.u1f, w.gbuf);
    gemm(stream, w.gbuf, mlp1, w.h1, BATCH, 960, 512, 0, w);
    bnrun(stream, w.h1, mbn1, w.h1, BATCH, 512, 1, nullptr, w);
    gemm(stream, w.h1, mlp2, w.h2, BATCH, 512, 256, 0, w);
    bnrun(stream, w.h2, mbn2, w.h2, BATCH, 256, 1, nullptr, w);
    gemm(stream, w.h2, mlp3, (float*)d_out, BATCH, 256, 6, 0, w);
}
